// HeadAttn_76527727280523
// MI455X (gfx1250) — matrix operand order repaired and checked by reading, not compiled, not run on hardware
//
#include <hip/hip_runtime.h>
#include <hip/hip_bf16.h>

// ---------- model constants ----------
#define NB      4
#define LSEQ    1024
#define DMODEL  256
#define NHEAD   8
#define DK      32
#define NLAYER  3
#define PASTW   12
#define FUTUREW 4
#define LPAST   (LSEQ - PASTW)        // 1012
#define LN_EPSF 1e-6f

#define RES_ELEMS  ((size_t)NB * LPAST * FUTUREW * DMODEL)          // 4,145,152
#define ATTN_ELEMS ((size_t)NB * NLAYER * NHEAD * LSEQ * LSEQ)      // 100,663,296

typedef __attribute__((ext_vector_type(16))) _Float16 v16h;
typedef __attribute__((ext_vector_type(8)))  float    v8f;
typedef __attribute__((ext_vector_type(4)))  float    f4;

// CDNA5 async global->LDS copy path (ASYNCcnt), with safe fallback
#if defined(__HIP_DEVICE_COMPILE__) &&                                   \
    __has_builtin(__builtin_amdgcn_global_load_async_to_lds_b32) &&      \
    __has_builtin(__builtin_amdgcn_s_wait_asynccnt)
#define HAVE_ASYNC_LDS 1
typedef __attribute__((address_space(1))) int g_as_int;   // global
typedef __attribute__((address_space(3))) int l_as_int;   // LDS
#else
#define HAVE_ASYNC_LDS 0
#endif

// Map compact output row -> activation row (skip first PASTW rows of each batch)
template<int ROWMODE>
__device__ __forceinline__ int map_row(int r) {
    if (ROWMODE == 0) return r;
    int b = r / LPAST;
    int t = r - b * LPAST;
    return b * LSEQ + PASTW + t;
}

// ============================================================================
// WMMA GEMM: C[M,N] = A[M,K] @ W[K,N] + bias[N] (+ residual epilogue)
//
// Block (8 waves) owns one 16-column N-tile: the K x 16 fp32 weight panel is
// staged once into LDS (async global->LDS when available), padded to a 21-float
// row stride so the two wave-halves gather B fragments from disjoint bank
// groups. Each wave computes MSUB stacked 16x16 tiles; the B fragment is read
// from LDS once per K-step and reused by MSUB WMMAs.
//
// ROWMODE: 0 -> A row == C row; 1 -> A row skips PASTW rows per batch.
// RESMODE: 0 none; 1 C += Rsd[row*N+col]; 2 C += Rsd[maprow(row)*256+col%256]
//          (RESMODE==2 streams C with nontemporal stores)
// Requires K <= 256 (K == DMODEL here).
// ============================================================================
template<int MSUB, int ROWMODE, int RESMODE>
__global__ __launch_bounds__(256)
void wmma_gemm_kernel(const float* __restrict__ A, const float* __restrict__ W,
                      const float* __restrict__ bias, const float* __restrict__ Rsd,
                      float* __restrict__ C, int M, int N, int K)
{
    constexpr int LDSTR = 21;                  // padded row stride (floats)
    __shared__ float ldsW[DMODEL * LDSTR];     // 21.5 KB weight panel

    const int tid  = threadIdx.x;
    const int lane = tid & 31;
    const int wv   = tid >> 5;                 // wave in block [0,8)
    const int tilesN     = N >> 4;
    const int rowsPerBlk = 8 * MSUB * 16;
    const int n0   = (blockIdx.x % tilesN) << 4;
    const int m0   = (blockIdx.x / tilesN) * rowsPerBlk + wv * (MSUB * 16);
    const int h    = lane >> 4;                // wave half (K-split)
    const int mr   = lane & 15;                // A row / B,C column within tile
    const int ncol = n0 + mr;

    // ---- stage W[:, n0:n0+16] into LDS (one panel per block) ----
    for (int e = tid; e < K * 16; e += 256) {
        const int k = e >> 4, n = e & 15;
        const float* gp = W + (size_t)k * N + (n0 + n);
#if HAVE_ASYNC_LDS
        __builtin_amdgcn_global_load_async_to_lds_b32(
            (g_as_int*)gp, (l_as_int*)&ldsW[k * LDSTR + n], 0, 0);
#else
        ldsW[k * LDSTR + n] = *gp;
#endif
    }
#if HAVE_ASYNC_LDS
    __builtin_amdgcn_s_wait_asynccnt(0);
#endif
    __syncthreads();

    // Per-subtile A row pointers (rows clamped for the padded last tile)
    const float* arp[MSUB];
#pragma unroll
    for (int st = 0; st < MSUB; ++st) {
        int r = m0 + st * 16 + mr;
        if (r >= M) r = M - 1;
        arp[st] = A + (size_t)map_row<ROWMODE>(r) * K;
    }

    v8f acc[MSUB];
#pragma unroll
    for (int st = 0; st < MSUB; ++st) { v8f z = {}; acc[st] = z; }

    for (int k0 = 0; k0 < K; k0 += 32) {
        // B fragment (32x16 f16) from LDS: slot j -> K row = k0 + 8h + j (j<8),
        // k0 + 16 + 8h + (j-8) (j>=8), column = mr. Stride-21 padding => halves use disjoint bank groups.
        v16h bf;
#pragma unroll
        for (int j = 0; j < 16; ++j)
            bf[j] = (_Float16)ldsW[(k0 + 8 * h + j + (j & 8)) * LDSTR + mr];

#pragma unroll
        for (int st = 0; st < MSUB; ++st) {
            // A fragment (16x32 f16): lane half h -> K = k0 + 8h + j (j<8),
            //                                       K = k0 + 16 + 8h + (j-8)
            const f4* ap  = reinterpret_cast<const f4*>(arp[st] + k0 + 8 * h);
            const f4* ap2 = reinterpret_cast<const f4*>(arp[st] + k0 + 16 + 8 * h);
            f4 a0 = ap[0], a1 = ap[1], a2 = ap2[0], a3 = ap2[1];
            v16h af;
            af[0]=(_Float16)a0.x;  af[1]=(_Float16)a0.y;  af[2]=(_Float16)a0.z;  af[3]=(_Float16)a0.w;
            af[4]=(_Float16)a1.x;  af[5]=(_Float16)a1.y;  af[6]=(_Float16)a1.z;  af[7]=(_Float16)a1.w;
            af[8]=(_Float16)a2.x;  af[9]=(_Float16)a2.y;  af[10]=(_Float16)a2.z; af[11]=(_Float16)a2.w;
            af[12]=(_Float16)a3.x; af[13]=(_Float16)a3.y; af[14]=(_Float16)a3.z; af[15]=(_Float16)a3.w;

            acc[st] = __builtin_amdgcn_wmma_f32_16x16x32_f16(
                /*neg_a=*/false, af, /*neg_b=*/false, bf,
                /*c_mod=*/(short)0, acc[st], /*reuse_a=*/false, /*reuse_b=*/false);
        }
    }

    // Epilogue: acc[st][r] = D[m0 + st*16 + r + 8h][ncol]
    const float bcol = bias[ncol];
#pragma unroll
    for (int st = 0; st < MSUB; ++st) {
#pragma unroll
        for (int r = 0; r < 8; ++r) {
            const int row = m0 + st * 16 + r + 8 * h;
            if (row < M) {
                float v = acc[st][r] + bcol;
                if (RESMODE == 1) {
                    v += Rsd[(size_t)row * N + ncol];
                    C[(size_t)row * N + ncol] = v;
                } else if (RESMODE == 2) {
                    const int ir = map_row<1>(row);
                    v += Rsd[(size_t)ir * DMODEL + (ncol & (DMODEL - 1))];
                    __builtin_nontemporal_store(v, C + (size_t)row * N + ncol);
                } else {
                    C[(size_t)row * N + ncol] = v;
                }
            }
        }
    }
}

// ============================================================================
// Banded attention: one wave per (b, head, query i). Band j in [max(0,i-11), i],
// at most 12 keys. Lane=key for scores/softmax (wave32 shfl reductions),
// lane=dim (dk=32) for the context accumulation. Writes only the band into the
// pre-zeroed attn tensor (nontemporal; write-once data).
// ============================================================================
__global__ __launch_bounds__(256)
void attn_kernel(const float* __restrict__ Q, const float* __restrict__ Km,
                 const float* __restrict__ V, float* __restrict__ attn,
                 float* __restrict__ ctx, int layer)
{
    const int lane = threadIdx.x & 31;
    const int w = blockIdx.x * (blockDim.x >> 5) + (threadIdx.x >> 5);
    const int b   = w / (NHEAD * LSEQ);
    const int rem = w % (NHEAD * LSEQ);
    const int hh  = rem / LSEQ;
    const int i   = rem % LSEQ;

    int jstart = i - (PASTW - 1); if (jstart < 0) jstart = 0;
    const int nj = i - jstart + 1;                 // <= 12
    const float scale = 0.17677669529663687f;      // 1/sqrt(32)

    float s = -3.4e38f;
    if (lane < nj) {
        const int j = jstart + lane;
        const float* qp = Q  + ((size_t)(b * LSEQ + i) * DMODEL + hh * DK);
        const float* kp = Km + ((size_t)(b * LSEQ + j) * DMODEL + hh * DK);
        float d = 0.f;
#pragma unroll
        for (int dd = 0; dd < DK; ++dd) d += qp[dd] * kp[dd];
        s = d * scale;
    }
    float m = s;
#pragma unroll
    for (int off = 16; off > 0; off >>= 1) m = fmaxf(m, __shfl_xor(m, off, 32));
    float p = (lane < nj) ? expf(s - m) : 0.f;
    float sum = p;
#pragma unroll
    for (int off = 16; off > 0; off >>= 1) sum += __shfl_xor(sum, off, 32);
    p = p / sum;

    if (lane < nj) {
        size_t off = ((((size_t)b * NLAYER + layer) * NHEAD + hh) * LSEQ + i) * LSEQ
                     + (size_t)(jstart + lane);
        __builtin_nontemporal_store(p, attn + off);
    }

    // ctx[d] = sum_j p_j * V[j][d], lane = d
    float acc = 0.f;
    for (int jj = 0; jj < nj; ++jj) {
        float pj = __shfl(p, jj, 32);
        acc += pj * V[(size_t)(b * LSEQ + jstart + jj) * DMODEL + hh * DK + lane];
    }
    ctx[(size_t)(b * LSEQ + i) * DMODEL + hh * DK + lane] = acc;
}

// ============================================================================
// LayerNorm: one wave per row of 256, 8 elements per lane (two float4 loads),
// wave32 shuffle reductions for mean / variance.
// ============================================================================
__global__ __launch_bounds__(256)
void layernorm_kernel(const float* __restrict__ Y, const float* __restrict__ gamma,
                      const float* __restrict__ beta, float* __restrict__ O)
{
    const int lane = threadIdx.x & 31;
    const int r = blockIdx.x * (blockDim.x >> 5) + (threadIdx.x >> 5);
    const f4* yp = reinterpret_cast<const f4*>(Y + (size_t)r * DMODEL + lane * 8);
    f4 v0 = yp[0], v1 = yp[1];
    float sm = v0.x + v0.y + v0.z + v0.w + v1.x + v1.y + v1.z + v1.w;
    float sq = v0.x*v0.x + v0.y*v0.y + v0.z*v0.z + v0.w*v0.w
             + v1.x*v1.x + v1.y*v1.y + v1.z*v1.z + v1.w*v1.w;
#pragma unroll
    for (int off = 16; off > 0; off >>= 1) {
        sm += __shfl_xor(sm, off, 32);
        sq += __shfl_xor(sq, off, 32);
    }
    const float mu  = sm * (1.f / DMODEL);
    const float var = sq * (1.f / DMODEL) - mu * mu;
    const float inv = rsqrtf(var + LN_EPSF);
    const int c0 = lane * 8;
    float* op = O + (size_t)r * DMODEL + c0;
    const float* gp = gamma + c0;
    const float* bp = beta + c0;
    float y[8] = {v0.x, v0.y, v0.z, v0.w, v1.x, v1.y, v1.z, v1.w};
#pragma unroll
    for (int e = 0; e < 8; ++e)
        op[e] = gp[e] * (y[e] - mu) * inv + bp[e];
}

// ============================================================================
// Streaming zero-fill (nontemporal float4 stores) for the huge attn output.
// ============================================================================
__global__ __launch_bounds__(256)
void zero_kernel(f4* __restrict__ p, size_t n4)
{
    size_t i = (size_t)blockIdx.x * blockDim.x + threadIdx.x;
    const size_t stride = (size_t)gridDim.x * blockDim.x;
    f4 z = {0.f, 0.f, 0.f, 0.f};
    for (; i < n4; i += stride)
        __builtin_nontemporal_store(z, p + i);
}

// ============================================================================
extern "C" void kernel_launch(void* const* d_in, const int* in_sizes, int n_in,
                              void* d_out, int out_size, void* d_ws, size_t ws_size,
                              hipStream_t stream)
{
    (void)in_sizes; (void)n_in; (void)out_size; (void)ws_size;

    const float* inp   = (const float*)d_in[0];
    const float* Wq    = (const float*)d_in[1];
    const float* bq    = (const float*)d_in[2];
    const float* Wk    = (const float*)d_in[3];
    const float* bk    = (const float*)d_in[4];
    const float* Wv    = (const float*)d_in[5];
    const float* bv    = (const float*)d_in[6];
    const float* Wo    = (const float*)d_in[7];
    const float* bo    = (const float*)d_in[8];
    const float* gamma = (const float*)d_in[9];
    const float* beta  = (const float*)d_in[10];
    const float* W_out = (const float*)d_in[11];
    const float* b_out = (const float*)d_in[12];

    float* res  = (float*)d_out;
    float* attn = res + RES_ELEMS;

    // workspace: 6 activation buffers of B*L*D floats (4 MB each, 24 MB total)
    const size_t AB = (size_t)NB * LSEQ * DMODEL;   // 1,048,576
    float* X  = (float*)d_ws;
    float* Qb = X  + AB;
    float* Kb = Qb + AB;
    float* Vb = Kb + AB;
    float* Cx = Vb + AB;
    float* Yb = Cx + AB;

    // 1) zero the (mostly-zero) attention tensor with streaming stores
    zero_kernel<<<8192, 256, 0, stream>>>((f4*)attn, ATTN_ELEMS / 4);

    const int Mg = NB * LSEQ;                        // 4096
    // MSUB=2: block = 8 waves * 32 rows = 256 rows x 16 cols
    const int gB = ((Mg + 255) / 256) * (DMODEL / 16);   // 16 * 16 = 256 blocks

    const size_t WSTR = (size_t)DMODEL * DMODEL;     // per-layer weight stride

    for (int l = 0; l < NLAYER; ++l) {
        const float* Xin = (l == 0) ? inp : X;
        wmma_gemm_kernel<2, 0, 0><<<gB, 256, 0, stream>>>(
            Xin, Wq + l * WSTR, bq + l * DMODEL, nullptr, Qb, Mg, DMODEL, DMODEL);
        wmma_gemm_kernel<2, 0, 0><<<gB, 256, 0, stream>>>(
            Xin, Wk + l * WSTR, bk + l * DMODEL, nullptr, Kb, Mg, DMODEL, DMODEL);
        wmma_gemm_kernel<2, 0, 0><<<gB, 256, 0, stream>>>(
            Xin, Wv + l * WSTR, bv + l * DMODEL, nullptr, Vb, Mg, DMODEL, DMODEL);
        attn_kernel<<<(NB * NHEAD * LSEQ) / 8, 256, 0, stream>>>(Qb, Kb, Vb, attn, Cx, l);
        wmma_gemm_kernel<2, 0, 1><<<gB, 256, 0, stream>>>(
            Cx, Wo + l * WSTR, bo + l * DMODEL, Xin, Yb, Mg, DMODEL, DMODEL);
        layernorm_kernel<<<Mg / 8, 256, 0, stream>>>(Yb, gamma + l * DMODEL,
                                                     beta + l * DMODEL, X);
    }

    // 2) final projection: out[:,PAST:] @ W_out + b_out + broadcast(inp[:,PAST:])
    //    M=4048 (padded to 512-row blocks, stores guarded), N=1024, MSUB=4
    const int Mf = NB * LPAST;                       // 4048
    const int Nf = FUTUREW * DMODEL;                 // 1024
    const int gF = ((Mf + 511) / 512) * (Nf / 16);   // 8 * 64 = 512 blocks
    wmma_gemm_kernel<4, 1, 2><<<gF, 256, 0, stream>>>(
        X, W_out, b_out, inp, res, Mf, Nf, DMODEL);
}
